// RendNet_51410758533390
// MI455X (gfx1250) — matrix operand order repaired and checked by reading, not compiled, not run on hardware
//
#include <hip/hip_runtime.h>
#include <hip/hip_bf16.h>
#include <math.h>

typedef __attribute__((ext_vector_type(16))) _Float16 v16h;
typedef __attribute__((ext_vector_type(8)))  float    v8f;

// ---------------------------------------------------------------------------
// Bilinear corner helper: matches _point_sample (zero padding, half-pixel).
// ---------------------------------------------------------------------------
__device__ __forceinline__ void bilinear_corners(float px, float py, int H, int W,
                                                 int* cidx, float* cw) {
  float x = px * (float)W - 0.5f;
  float y = py * (float)H - 0.5f;
  float x0f = floorf(x), y0f = floorf(y);
  float wx = x - x0f, wy = y - y0f;
  int x0 = (int)x0f, y0 = (int)y0f;
  int   xs[2]  = {x0, x0 + 1};
  int   ys[2]  = {y0, y0 + 1};
  float wxs[2] = {1.f - wx, wx};
  float wys[2] = {1.f - wy, wy};
#pragma unroll
  for (int cy = 0; cy < 2; ++cy)
#pragma unroll
    for (int cx = 0; cx < 2; ++cx) {
      int xi = xs[cx], yi = ys[cy];
      bool valid = (xi >= 0) && (xi < W) && (yi >= 0) && (yi < H);
      int xc = xi < 0 ? 0 : (xi > W - 1 ? W - 1 : xi);
      int yc = yi < 0 ? 0 : (yi > H - 1 ? H - 1 : yi);
      cidx[cy * 2 + cx] = yc * W + xc;
      cw[cy * 2 + cx]   = valid ? wxs[cx] * wys[cy] : 0.f;
    }
}

// ---------------------------------------------------------------------------
// 2x bilinear upsample (align_corners=False / half-pixel, edge clamp).
// ---------------------------------------------------------------------------
__global__ void resize2x_kernel(const float* __restrict__ src, float* __restrict__ dst,
                                int BC, int H, int W) {
  int Ho = 2 * H, Wo = 2 * W;
  int t = blockIdx.x * blockDim.x + threadIdx.x;
  int total = BC * Ho * Wo;
  if (t >= total) return;
  int wo = t % Wo;
  int ho = (t / Wo) % Ho;
  int bc = t / (Wo * Ho);
  float sx = (wo + 0.5f) * 0.5f - 0.5f;
  float sy = (ho + 0.5f) * 0.5f - 0.5f;
  float x0f = floorf(sx), y0f = floorf(sy);
  float fx = sx - x0f, fy = sy - y0f;
  int x0 = (int)x0f, y0 = (int)y0f;
  int x0c = x0 < 0 ? 0 : (x0 > W - 1 ? W - 1 : x0);
  int x1c = (x0 + 1) < 0 ? 0 : ((x0 + 1) > W - 1 ? W - 1 : x0 + 1);
  int y0c = y0 < 0 ? 0 : (y0 > H - 1 ? H - 1 : y0);
  int y1c = (y0 + 1) < 0 ? 0 : ((y0 + 1) > H - 1 ? H - 1 : y0 + 1);
  const float* s = src + (size_t)bc * H * W;
  float v = (1.f - fy) * ((1.f - fx) * s[y0c * W + x0c] + fx * s[y0c * W + x1c]) +
            fy * ((1.f - fx) * s[y1c * W + x0c] + fx * s[y1c * W + x1c]);
  dst[t] = v;
}

// ---------------------------------------------------------------------------
// Uncertainty: sample softmax(temp) at oversample points, unc = -(top1-top2).
// ---------------------------------------------------------------------------
__global__ void unc_kernel(const float* __restrict__ temp, const float* __restrict__ over,
                           float* __restrict__ unc, int B, int kN, int H, int W) {
  int t = blockIdx.x * blockDim.x + threadIdx.x;
  if (t >= B * kN) return;
  int b = t / kN, j = t % kN;
  float px = over[((size_t)b * kN + j) * 2 + 0];
  float py = over[((size_t)b * kN + j) * 2 + 1];
  int cidx[4]; float cw[4];
  bilinear_corners(px, py, H, W, cidx, cw);
  const float* tb = temp + (size_t)b * 3 * H * W;
  int HW = H * W;
  float v0 = 0.f, v1 = 0.f, v2 = 0.f;
#pragma unroll
  for (int c = 0; c < 4; ++c) {
    float t0 = tb[cidx[c]];
    float t1 = tb[HW + cidx[c]];
    float t2 = tb[2 * HW + cidx[c]];
    float mx = fmaxf(t0, fmaxf(t1, t2));
    float e0 = __expf(t0 - mx), e1 = __expf(t1 - mx), e2 = __expf(t2 - mx);
    float inv = 1.f / (e0 + e1 + e2);
    v0 += cw[c] * e0 * inv;
    v1 += cw[c] * e1 * inv;
    v2 += cw[c] * e2 * inv;
  }
  float m1 = fmaxf(v0, fmaxf(v1, v2));
  float mn = fminf(v0, fminf(v1, v2));
  float m2 = v0 + v1 + v2 - m1 - mn;
  unc[t] = m2 - m1;  // = -(top1 - top2)
}

// ---------------------------------------------------------------------------
// Iterative top-k (one block per batch). Writes [imp; cov] points to pts_out.
// Destroys unc (rebuilt each call). Tie-break: smallest index wins.
// ---------------------------------------------------------------------------
__global__ void topk_kernel(float* __restrict__ unc, const float* __restrict__ over,
                            const float* __restrict__ cov, float* __restrict__ pts_out,
                            int kN, int n_imp, int ncov) {
  int b = blockIdx.x;
  float* u = unc + (size_t)b * kN;
  const float* ov = over + (size_t)b * kN * 2;
  int N = n_imp + ncov;
  __shared__ float sv[256];
  __shared__ int   si[256];
  for (int it = 0; it < n_imp; ++it) {
    float bv = -3.4e38f;
    int bi = 0x7fffffff;
    for (int i = threadIdx.x; i < kN; i += blockDim.x) {
      float v = u[i];
      if (v > bv || (v == bv && i < bi)) { bv = v; bi = i; }
    }
    sv[threadIdx.x] = bv;
    si[threadIdx.x] = bi;
    __syncthreads();
    for (int s = blockDim.x / 2; s > 0; s >>= 1) {
      if ((int)threadIdx.x < s) {
        float v2 = sv[threadIdx.x + s];
        int i2 = si[threadIdx.x + s];
        if (v2 > sv[threadIdx.x] || (v2 == sv[threadIdx.x] && i2 < si[threadIdx.x])) {
          sv[threadIdx.x] = v2;
          si[threadIdx.x] = i2;
        }
      }
      __syncthreads();
    }
    if (threadIdx.x == 0) {
      int idx = si[0];
      pts_out[((size_t)b * N + it) * 2 + 0] = ov[(size_t)idx * 2 + 0];
      pts_out[((size_t)b * N + it) * 2 + 1] = ov[(size_t)idx * 2 + 1];
      u[idx] = -3.4e38f;
    }
    __syncthreads();
  }
  for (int j = threadIdx.x; j < ncov; j += blockDim.x) {
    pts_out[((size_t)b * N + n_imp + j) * 2 + 0] = cov[((size_t)b * ncov + j) * 2 + 0];
    pts_out[((size_t)b * N + n_imp + j) * 2 + 1] = cov[((size_t)b * ncov + j) * 2 + 1];
  }
}

// ---------------------------------------------------------------------------
// Gather [cf; ff] per point into transposed f16 matrix Xt[B*N][Kpad], zero-pad.
// ---------------------------------------------------------------------------
__global__ void gather_kernel(const float* __restrict__ temp, int Ht, int Wt,
                              const float* __restrict__ feat, int C, int Hf, int Wf,
                              const float* __restrict__ pts, int N, int Kpad,
                              _Float16* __restrict__ Xt) {
  int n = blockIdx.x;  // 0..B*N-1
  int b = n / N, j = n % N;
  float px = pts[((size_t)b * N + j) * 2 + 0];
  float py = pts[((size_t)b * N + j) * 2 + 1];
  int ci_t[4], ci_f[4];
  float cw_t[4], cw_f[4];
  bilinear_corners(px, py, Ht, Wt, ci_t, cw_t);
  bilinear_corners(px, py, Hf, Wf, ci_f, cw_f);
  const float* tb = temp + (size_t)b * 3 * Ht * Wt;
  const float* fb = feat + (size_t)b * C * Hf * Wf;
  for (int c = threadIdx.x; c < Kpad; c += blockDim.x) {
    float v = 0.f;
    if (c < 3) {
      const float* ch = tb + (size_t)c * Ht * Wt;
#pragma unroll
      for (int k = 0; k < 4; ++k) v += cw_t[k] * ch[ci_t[k]];
    } else if (c < 3 + C) {
      const float* ch = fb + (size_t)(c - 3) * Hf * Wf;
#pragma unroll
      for (int k = 0; k < 4; ++k) v += cw_f[k] * ch[ci_f[k]];
    }
    Xt[(size_t)n * Kpad + c] = (_Float16)v;
  }
}

// ---------------------------------------------------------------------------
// f32 weights -> padded f16 [Opad][Kpad].
// ---------------------------------------------------------------------------
__global__ void wcvt_kernel(const float* __restrict__ W, int Oreal, int Kin,
                            int Opad, int Kpad, _Float16* __restrict__ Wp) {
  int t = blockIdx.x * blockDim.x + threadIdx.x;
  if (t >= Opad * Kpad) return;
  int o = t / Kpad, k = t % Kpad;
  float v = (o < Oreal && k < Kin) ? W[(size_t)o * Kin + k] : 0.f;
  Wp[t] = (_Float16)v;
}

// ---------------------------------------------------------------------------
// WMMA GEMM: Y = W(Opad x Kpad) * Xt^T(Kpad x Nt) + bias, one wave per 16x16
// tile of Y, K in steps of 32 via v_wmma_f32_16x16x32_f16.
//   mode 0: Hout[n*256 + m] = ReLU(y) as f16 (hidden layer, Oreal==256)
//   mode 1: rend[b*3*Nper + m*Nper + j] = y  for m < Oreal(=3)
// A layout (16x32 f16, M x K):  lane = {m = lane&15}; VGPR v<4: K = 2v + 8*(lane>=16),
//   v>=4: K = 2v + 8 + 8*(lane>=16)    (ISA 7.12.2)
// B layout (32x16 f16, K x N):  lane = {n = lane&15}; same K mapping as A
// C/D (16x16 f32): lane n = lane&15, VGPR r: m = r + 8*(lane>=16)
// ---------------------------------------------------------------------------
__global__ void gemm_wmma_kernel(const _Float16* __restrict__ Wp,
                                 const _Float16* __restrict__ Xt,
                                 const float* __restrict__ bias, int Oreal,
                                 int Kpad, int mode,
                                 _Float16* __restrict__ Hout,
                                 float* __restrict__ rend, int Nper) {
  int tn = blockIdx.x;  // N tile
  int tm = blockIdx.y;  // M tile
  int lane = threadIdx.x;
  int lh = lane >> 4;
  int lm = lane & 15;
  v8f acc = {};
  const _Float16* wrow = Wp + (size_t)(tm * 16 + lm) * Kpad;
  const _Float16* xrow = Xt + (size_t)(tn * 16 + lm) * Kpad;
  for (int kk = 0; kk < Kpad; kk += 32) {
    v16h a, bb;
#pragma unroll
    for (int v = 0; v < 8; ++v) {
      int ka = ((v < 4) ? 2 * v : 2 * v + 8) + (lh ? 8 : 0);
      a[2 * v]     = wrow[kk + ka];
      a[2 * v + 1] = wrow[kk + ka + 1];
      int kb = ((v < 4) ? 2 * v : 2 * v + 8) + (lh ? 8 : 0);
      bb[2 * v]     = xrow[kk + kb];
      bb[2 * v + 1] = xrow[kk + kb + 1];
    }
    acc = __builtin_amdgcn_wmma_f32_16x16x32_f16(
        /*neg_a=*/false, a, /*neg_b=*/false, bb,
        /*c_mod=*/(short)0, acc, /*reuse_a=*/false, /*reuse_b=*/false);
  }
  int n = tn * 16 + lm;
#pragma unroll
  for (int r = 0; r < 8; ++r) {
    int m = tm * 16 + r + 8 * lh;
    float v = acc[r];
    if (m < Oreal) v += bias[m];
    if (mode == 0) {
      v = fmaxf(v, 0.f);
      Hout[(size_t)n * 256 + m] = (_Float16)v;
    } else if (m < Oreal) {
      int bidx = n / Nper, j = n % Nper;
      rend[(size_t)bidx * 3 * Nper + (size_t)m * Nper + j] = v;
    }
  }
}

// ---------------------------------------------------------------------------
// Host orchestration
// ---------------------------------------------------------------------------
extern "C" void kernel_launch(void* const* d_in, const int* in_sizes, int n_in,
                              void* d_out, int out_size, void* d_ws, size_t ws_size,
                              hipStream_t stream) {
  (void)in_sizes; (void)n_in; (void)out_size; (void)ws_size;
  const int B = 4;
  const float* refine = (const float*)d_in[0];   // [4,64,512,512]
  const float* x0     = (const float*)d_in[1];   // [4,64,256,256]
  const float* x1     = (const float*)d_in[2];   // [4,256,128,128]
  const float* coarse = (const float*)d_in[5];   // [4,3,64,64]
  const float* over3  = (const float*)d_in[6];
  const float* cov3   = (const float*)d_in[7];
  const float* over4  = (const float*)d_in[8];
  const float* cov4   = (const float*)d_in[9];
  const float* over5  = (const float*)d_in[10];
  const float* cov5   = (const float*)d_in[11];
  // MLP params: m1(12..19), m0(20..27), mr(28..35): w1,b1,w2,b2,w3,b3,wf,bf
  const float* mw[3][4];
  const float* mb[3][4];
  for (int g = 0; g < 3; ++g)
    for (int l = 0; l < 4; ++l) {
      mw[g][l] = (const float*)d_in[12 + g * 8 + l * 2];
      mb[g][l] = (const float*)d_in[12 + g * 8 + l * 2 + 1];
    }

  float* out = (float*)d_out;
  const size_t o_p3 = 49152, o_r3 = 50176, o_p4 = 51712, o_r4 = 53760,
               o_p5 = 56832, o_r5 = 60928;

  // ---- workspace bump allocator ----
  char* ws = (char*)d_ws;
  size_t off = 0;
  auto walloc = [&](size_t bytes) -> void* {
    void* p = ws + off;
    off = (off + bytes + 255) & ~(size_t)255;
    return p;
  };
  float* temp3 = (float*)walloc((size_t)B * 3 * 128 * 128 * 4);
  float* temp4 = (float*)walloc((size_t)B * 3 * 256 * 256 * 4);
  float* temp5 = (float*)walloc((size_t)B * 3 * 512 * 512 * 4);
  float* unc   = (float*)walloc((size_t)B * 102400 * 4);
  _Float16* Xt = (_Float16*)walloc((size_t)2048 * 288 * 2);
  _Float16* H1 = (_Float16*)walloc((size_t)2048 * 256 * 2);
  _Float16* H2 = (_Float16*)walloc((size_t)2048 * 256 * 2);
  // padded f16 weights: per group, Kpad1 = 288 (m1) or 96 (m0/mr)
  _Float16* Wp[3][4];
  int Kp1[3] = {288, 96, 96};
  for (int g = 0; g < 3; ++g) {
    Wp[g][0] = (_Float16*)walloc((size_t)256 * Kp1[g] * 2);
    Wp[g][1] = (_Float16*)walloc((size_t)256 * 256 * 2);
    Wp[g][2] = (_Float16*)walloc((size_t)256 * 256 * 2);
    Wp[g][3] = (_Float16*)walloc((size_t)16 * 256 * 2);
  }

  // ---- pass-through output: coarse ----
  hipMemcpyAsync(out, coarse, (size_t)B * 3 * 64 * 64 * 4,
                 hipMemcpyDeviceToDevice, stream);

  // ---- weight conversion (independent of stages) ----
  int Kin1[3] = {259, 67, 67};
  for (int g = 0; g < 3; ++g) {
    int dims[4][4] = {{256, Kin1[g], 256, Kp1[g]},
                      {256, 256, 256, 256},
                      {256, 256, 256, 256},
                      {3, 256, 16, 256}};
    for (int l = 0; l < 4; ++l) {
      int total = dims[l][2] * dims[l][3];
      wcvt_kernel<<<(total + 255) / 256, 256, 0, stream>>>(
          mw[g][l], dims[l][0], dims[l][1], dims[l][2], dims[l][3], Wp[g][l]);
    }
  }

  // ---- temp chain ----
  {
    int t3 = B * 3 * 128 * 128;
    resize2x_kernel<<<(t3 + 255) / 256, 256, 0, stream>>>(coarse, temp3, B * 3, 64, 64);
    int t4 = B * 3 * 256 * 256;
    resize2x_kernel<<<(t4 + 255) / 256, 256, 0, stream>>>(temp3, temp4, B * 3, 128, 128);
    int t5 = B * 3 * 512 * 512;
    resize2x_kernel<<<(t5 + 255) / 256, 256, 0, stream>>>(temp4, temp5, B * 3, 256, 256);
  }

  // ---- generic stage runner ----
  auto run_stage = [&](const float* temp, int Ht, int Wt,
                       const float* feat, int C, int Hf, int Wf,
                       const float* over, int kN, const float* cov, int n_imp,
                       int ncov, int g, int Kpad, float* pts_out, float* rend_out) {
    int N = n_imp + ncov;       // points per batch
    int Nt = B * N;             // total GEMM columns (multiple of 16)
    unc_kernel<<<(B * kN + 255) / 256, 256, 0, stream>>>(temp, over, unc, B, kN, Ht, Wt);
    topk_kernel<<<B, 256, 0, stream>>>(unc, over, cov, pts_out, kN, n_imp, ncov);
    gather_kernel<<<Nt, 128, 0, stream>>>(temp, Ht, Wt, feat, C, Hf, Wf,
                                          pts_out, N, Kpad, Xt);
    dim3 wblk(32);
    gemm_wmma_kernel<<<dim3(Nt / 16, 16), wblk, 0, stream>>>(
        Wp[g][0], Xt, mb[g][0], 256, Kpad, 0, H1, nullptr, N);
    gemm_wmma_kernel<<<dim3(Nt / 16, 16), wblk, 0, stream>>>(
        Wp[g][1], H1, mb[g][1], 256, 256, 0, H2, nullptr, N);
    gemm_wmma_kernel<<<dim3(Nt / 16, 16), wblk, 0, stream>>>(
        Wp[g][2], H2, mb[g][2], 256, 256, 0, H1, nullptr, N);
    gemm_wmma_kernel<<<dim3(Nt / 16, 1), wblk, 0, stream>>>(
        Wp[g][3], H1, mb[g][3], 3, 256, 1, nullptr, rend_out, N);
  };

  // stage 3: temp3 [128x128], feat x1 (C=256,128x128), 121+7 points, Kpad 288
  run_stage(temp3, 128, 128, x1, 256, 128, 128, over3, 200 * 128, cov3, 121, 7,
            0, 288, out + o_p3, out + o_r3);
  // stage 4: temp4 [256x256], feat x0 (C=64,256x256), 243+13 points, Kpad 96
  run_stage(temp4, 256, 256, x0, 64, 256, 256, over4, 200 * 256, cov4, 243, 13,
            1, 96, out + o_p4, out + o_r4);
  // stage 5: temp5 [512x512], feat refine (C=64,512x512), 486+26 points, Kpad 96
  run_stage(temp5, 512, 512, refine, 64, 512, 512, over5, 200 * 512, cov5, 486, 26,
            2, 96, out + o_p5, out + o_r5);
}
